// PrimaryCaps_23871428231768
// MI455X (gfx1250) — matrix operand order repaired and checked by reading, not compiled, not run on hardware
//
#include <hip/hip_runtime.h>

// ---------------------------------------------------------------------------
// PrimaryCaps as implicit GEMM on CDNA5 WMMA (bf16 -> f32 accumulate)
//   M = 128*28*28 = 100352 output pixels, N = 392 channels, K = 16*81 = 1296
//   Kpad = 1312 (41 x 32), Npad = 400 (25 x 16 = 5 waves x 5 tiles)
// A k-step covers exactly two 9x9 taps (C_IN==16), so every A and B fragment
// is two aligned b128 loads. Ping-pong double buffering, branch-free k-loop,
// LDS tap-offset table, fused bias+squash epilogue via ds_swizzle.
// ---------------------------------------------------------------------------

typedef __attribute__((ext_vector_type(16))) __bf16 bf16x16;
typedef __attribute__((ext_vector_type(8)))  float  f32x8;

#define B_IMG   128
#define C_IN    16
#define HIN     64
#define WIN     64
#define HOUT    28
#define WOUT    28
#define NCH     392          // 49 units * 8 dims
#define NPAD    400
#define KDIM    1296         // 81 * 16
#define KSTEPS  41           // ceil(1296/32)
#define NTAPS   82           // 2*KSTEPS taps (tap 81 = benign zero-weight tail)
#define PIXPI   (HOUT*WOUT)  // 784 (== 49*16: M-tiles never straddle images)
#define MTILES  ((B_IMG*PIXPI)/16)   // 6272
#define NT_PER_WAVE 5
#define WAVES   5
#define BLOCK   (WAVES*32)   // 160
#define BSTEP   (2*NPAD*32)  // 25600 B: Wq advance per k-step

__device__ __forceinline__ unsigned short f2bf(float f) {
    unsigned u = __float_as_uint(f);
    u += 0x7FFFu + ((u >> 16) & 1u);
    return (unsigned short)(u >> 16);
}

// x[n][ci][iy][ix] (f32 NCHW) -> xT[n][iy][ix][ci] (bf16 NHWC):
// each half-wave A fragment becomes two contiguous 16B loads.
__global__ void pack_x_bf16(const float* __restrict__ x,
                            unsigned short* __restrict__ xT) {
    int idx = blockIdx.x * blockDim.x + threadIdx.x;
    if (idx >= B_IMG * HIN * WIN * C_IN) return;
    int ci = idx & 15;
    int ix = (idx >> 4) & 63;
    int iy = (idx >> 10) & 63;
    int n  = idx >> 16;
    float v = x[(((n * C_IN + ci) << 6) + iy) * 64 + ix];
    xT[idx] = f2bf(v);
}

// W[c][ci][ky][kx] -> Wq fragment-major: Wq[(g*NPAD + c)*8 + v] holds the
// bf16 pair for K-rows kp = 8g+v (g = 2s+i for b128 load i; h picks v = 4h..4h+3). A B fragment = 2 contiguous
// b128 loads; adjacent lanes (c) read adjacent 32B blocks. Zero-padded for
// c >= 392 and k >= 1296 (annihilates the in-bounds garbage K-tail A reads).
__global__ void pack_w_bf16(const float* __restrict__ W,
                            unsigned int* __restrict__ Wq) {
    int q = blockIdx.x * blockDim.x + threadIdx.x;
    if (q >= NTAPS * NPAD * 8) return;
    int v  = q & 7;
    int gc = q >> 3;
    int c  = gc % NPAD;
    int g  = gc / NPAD;
    int k  = (g * 8 + v) * 2;
    unsigned packed = 0u;
    if (k < KDIM && c < NCH) {
        int ci = k & 15;
        int kk = k >> 4;          // ky*9+kx
        float lo = W[(c * C_IN + ci    ) * 81 + kk];
        float hi = W[(c * C_IN + ci + 1) * 81 + kk];
        packed = (unsigned)f2bf(lo) | ((unsigned)f2bf(hi) << 16);
    }
    Wq[q] = packed;
}

// ---- fragment load/compute helpers (constant-folded under full unroll) ----
__device__ __forceinline__ void load_step(uint4* A, uint4 (*B)[2],
                                          const char* aPtr,
                                          const int* __restrict__ tap,
                                          const char* const* bp, int s) {
    int t0 = tap[2 * s];
    int t1 = tap[2 * s + 1];
    A[0] = *(const uint4*)(aPtr + t0);          // tap 2s,   ci = h*8..h*8+7
    A[1] = *(const uint4*)(aPtr + t1);          // tap 2s+1, ci = h*8..h*8+7
    const size_t off = (size_t)s * BSTEP;
#pragma unroll
    for (int j = 0; j < NT_PER_WAVE; ++j) {
        B[j][0] = *(const uint4*)(bp[j] + off);
        B[j][1] = *(const uint4*)(bp[j] + off + NPAD * 32);
    }
}

__device__ __forceinline__ f32x8 mma(const uint4* A, const uint4* Bj, f32x8 c) {
    union { uint4 q[2]; bf16x16 v; } a, b;
    a.q[0] = A[0]; a.q[1] = A[1];
    b.q[0] = Bj[0]; b.q[1] = Bj[1];
    return __builtin_amdgcn_wmma_f32_16x16x32_bf16(
        false, a.v, false, b.v, (short)0, c, false, false);
}

// One block = 5 waves, one 16-row M tile; wave w owns N-tiles [5w, 5w+5).
__global__ void __launch_bounds__(BLOCK)
caps_gemm(const unsigned short* __restrict__ xT,
          const unsigned int*  __restrict__ Wq,
          const float* __restrict__ bias,
          float* __restrict__ out) {
    __shared__ int tapOff[NTAPS];   // byte offset of each conv tap in xT

    for (int i = threadIdx.x; i < NTAPS; i += BLOCK) {
        int ky = i / 9;             // tap 81 -> (9,0): in-bounds, zero weights
        int kx = i - 9 * ky;
        tapOff[i] = (ky * WIN + kx) * C_IN * 2;
    }
    __syncthreads();

    const int lane = threadIdx.x & 31;
    const int wid  = __builtin_amdgcn_readfirstlane(threadIdx.x >> 5);
    const int h    = lane >> 4;
    const int ln   = lane & 15;
    const int h8   = h * 8;

    // A-side pixel for this lane
    const int n       = blockIdx.x / 49;
    const int remBase = (blockIdx.x % 49) * 16;
    const int remA    = remBase + ln;
    const int oy = remA / WOUT;
    const int ox = remA - oy * WOUT;
    const char* aPtr = (const char*)xT +
        ((size_t)(((unsigned)n * HIN + 2u * oy) * WIN + 2u * ox) * C_IN) * 2 +
        h * 16;                                    // ci = h*8 within each tap

    // per-tile weight fragment base: (g=2s)*NPAD + c, 32 B per column, +16 B for h
    const char* bp[NT_PER_WAVE];
#pragma unroll
    for (int j = 0; j < NT_PER_WAVE; ++j) {
        const int c = (wid * NT_PER_WAVE + j) * 16 + ln;
        bp[j] = (const char*)Wq + (size_t)c * 32 + h * 16;
    }

    f32x8 acc[NT_PER_WAVE];
#pragma unroll
    for (int j = 0; j < NT_PER_WAVE; ++j) acc[j] = (f32x8){0,0,0,0,0,0,0,0};

    // ---- software-pipelined k-loop: ping-pong buffers, no branches ----
    uint4 A0[2], A1[2], B0[NT_PER_WAVE][2], B1[NT_PER_WAVE][2];

    load_step(A0, B0, aPtr, tapOff, bp, 0);

    for (int s = 0; s + 2 <= KSTEPS; s += 2) {
        load_step(A1, B1, aPtr, tapOff, bp, s + 1);
#pragma unroll
        for (int j = 0; j < NT_PER_WAVE; ++j) acc[j] = mma(A0, B0[j], acc[j]);

        load_step(A0, B0, aPtr, tapOff, bp, s + 2);
#pragma unroll
        for (int j = 0; j < NT_PER_WAVE; ++j) acc[j] = mma(A1, B1[j], acc[j]);
    }
    // KSTEPS odd: final step already resident in buffer 0
#pragma unroll
    for (int j = 0; j < NT_PER_WAVE; ++j) acc[j] = mma(A0, B0[j], acc[j]);

    // ---- epilogue: bias + squash + capsule-ordered scatter ----
#pragma unroll
    for (int j = 0; j < NT_PER_WAVE; ++j) {
        const int c = (wid * NT_PER_WAVE + j) * 16 + ln;   // channel
        const float bc = (c < NCH) ? bias[c] : 0.0f;
        const int unit = c >> 3;
        const int dim  = c & 7;
#pragma unroll
        for (int r = 0; r < 8; ++r) {
            float val = acc[j][r] + bc;
            float p = val * val;
            // capsule = 8 consecutive lanes of this VGPR (xor 1,2,4 swizzles)
            p += __int_as_float(__builtin_amdgcn_ds_swizzle(__float_as_int(p), 0x041F));
            p += __int_as_float(__builtin_amdgcn_ds_swizzle(__float_as_int(p), 0x081F));
            p += __int_as_float(__builtin_amdgcn_ds_swizzle(__float_as_int(p), 0x101F));
            float mag = __builtin_sqrtf(p);
            float o   = val * (mag / (1.0f + p));          // squash
            if (c < NCH) {
                const int hw = remBase + r + h8;           // D-matrix M row
                out[((size_t)n * (PIXPI * 49) + (size_t)hw * 49 + unit) * 8 + dim] = o;
            }
        }
    }
}

extern "C" void kernel_launch(void* const* d_in, const int* in_sizes, int n_in,
                              void* d_out, int out_size, void* d_ws, size_t ws_size,
                              hipStream_t stream) {
    const float* x = (const float*)d_in[0];
    const float* W = (const float*)d_in[1];
    const float* b = (const float*)d_in[2];
    float* out = (float*)d_out;

    // workspace: xT (16 MiB bf16 NHWC) | Wq (1.0 MiB fragment-major bf16 pairs)
    unsigned short* xT = (unsigned short*)d_ws;
    unsigned int*   Wq = (unsigned int*)((char*)d_ws +
                           (size_t)B_IMG * HIN * WIN * C_IN * sizeof(unsigned short));

    const int nx = B_IMG * HIN * WIN * C_IN;           // 8,388,608
    hipLaunchKernelGGL(pack_x_bf16, dim3((nx + 255) / 256), dim3(256), 0, stream, x, xT);

    const int nw = NTAPS * NPAD * 8;                   // 262,400
    hipLaunchKernelGGL(pack_w_bf16, dim3((nw + 255) / 256), dim3(256), 0, stream, W, Wq);

    hipLaunchKernelGGL(caps_gemm, dim3(MTILES), dim3(BLOCK), 0, stream, xT, Wq, b, out);
}
